// RGBuvHistBlock_88192858456503
// MI455X (gfx1250) — matrix operand order repaired and checked by reading, not compiled, not run on hardware
//
#include <hip/hip_runtime.h>
#include <hip/hip_bf16.h>

// Problem constants (from reference): x[8,3,256,256] fp32 -> hists[8,3,64,64] fp32
#define L_SAMP 8
#define CHN    3
#define HB     64
#define NPIX   65536            // 256*256 pixels per (l,c)
#define SPLITS 16               // pixel-range splits per (l,c) for parallelism
#define CHUNK  128              // pixels staged in LDS per iteration (4 WMMA K-steps)
#define AROW   (CHUNK + 8)      // padded LDS row stride (halfs) to spread banks
#define PIX_PER_SPLIT (NPIX / SPLITS)
#define CHUNKS (PIX_PER_SPLIT / CHUNK)

typedef _Float16 v16h __attribute__((ext_vector_type(16)));
typedef _Float16 v8h  __attribute__((ext_vector_type(8)));
typedef float    v8f  __attribute__((ext_vector_type(8)));

__device__ __forceinline__ float fast_ln(float x) {
  // v_log_f32 is log2; scale to natural log
  return 0.69314718056f * __builtin_amdgcn_logf(x);
}

// ---------------------------------------------------------------------------
// Main kernel: per (l, c, split) workgroup builds a partial 64x64 histogram
// via f16 WMMA outer products and atomically accumulates into d_out.
// ---------------------------------------------------------------------------
__global__ __launch_bounds__(256, 2)
void hist_wmma_kernel(const float* __restrict__ x, float* __restrict__ out) {
  __shared__ _Float16 aLDS[HB * AROW];  // A = Iy * Ku, layout [u_bin][pixel]
  __shared__ _Float16 bLDS[HB * AROW];  // B = Kv,      layout [v_bin][pixel]
  __shared__ float uS[CHUNK];
  __shared__ float vS[CHUNK];
  __shared__ float iyS[CHUNK];

  const int split = blockIdx.x;
  const int c     = blockIdx.y;
  const int l     = blockIdx.z;
  const int tid   = threadIdx.x;
  const int lane  = tid & 31;           // wave32
  const int wave  = tid >> 5;           // 8 waves per block

  const float inv_s2 = 2500.0f;         // 1 / (0.02^2)
  const float EPSf   = 1e-6f;

  const float* __restrict__ xr = x + (size_t)l * CHN * NPIX;
  const int pbase = split * PIX_PER_SPLIT;

  // Wave -> output-tile mapping: 64x64 = 4x4 tiles of 16x16.
  // Wave w owns u-tile (w>>1) and v-tiles {2*(w&1), 2*(w&1)+1}.
  const int uT  = wave >> 1;
  const int vT0 = (wave & 1) << 1;

  v8f acc0 = {};
  v8f acc1 = {};

  for (int ck = 0; ck < CHUNKS; ++ck) {
    const int pix0 = pbase + ck * CHUNK;

    // ---- Phase 1: per-pixel attributes (u, v, Iy) -> LDS ----
    if (tid < CHUNK) {
      const int n = pix0 + tid;
      float r = xr[n];
      float g = xr[NPIX + n];
      float b = xr[2 * NPIX + n];
      r = fminf(fmaxf(r, 0.0f), 1.0f);
      g = fminf(fmaxf(g, 0.0f), 1.0f);
      b = fminf(fmaxf(b, 0.0f), 1.0f);
      const float lr = fast_ln(r + EPSf);
      const float lg = fast_ln(g + EPSf);
      const float lb = fast_ln(b + EPSf);
      float uu, vv;
      if (c == 0)      { uu = lr - lg; vv = lr - lb; }   // cu=1, cv=2
      else if (c == 1) { uu = lg - lr; vv = lg - lb; }   // cu=0, cv=2
      else             { uu = lb - lr; vv = lb - lg; }   // cu=0, cv=1
      uS[tid]  = uu;
      vS[tid]  = vv;
      iyS[tid] = __builtin_amdgcn_sqrtf(
          __builtin_fmaf(r, r, __builtin_fmaf(g, g, __builtin_fmaf(b, b, EPSf))));
    }
    __syncthreads();

    // ---- Phase 2: soft-binning kernels -> f16 LDS tiles (packed dwords) ----
    {
      const int kp = tid & 63;          // pixel pair index (2 pixels per thread)
      const int u0 = (tid >> 6) << 4;   // 16 bins per thread
      const float ua = uS[2 * kp],  ub = uS[2 * kp + 1];
      const float va = vS[2 * kp],  vb = vS[2 * kp + 1];
      const float ia = iyS[2 * kp], ib = iyS[2 * kp + 1];
#pragma unroll
      for (int j = 0; j < 16; ++j) {
        const int u = u0 + j;
        const float off = __builtin_fmaf((float)u, 6.0f / 63.0f, -3.0f);
        float d0 = ua - off, d1 = ub - off;
        const float ka = ia * __builtin_amdgcn_rcpf(__builtin_fmaf(d0 * d0, inv_s2, 1.0f));
        const float kb = ib * __builtin_amdgcn_rcpf(__builtin_fmaf(d1 * d1, inv_s2, 1.0f));
        d0 = va - off; d1 = vb - off;
        const float kc = __builtin_amdgcn_rcpf(__builtin_fmaf(d0 * d0, inv_s2, 1.0f));
        const float kd = __builtin_amdgcn_rcpf(__builtin_fmaf(d1 * d1, inv_s2, 1.0f));
        union { _Float16 h[2]; unsigned int w; } pk;
        pk.h[0] = (_Float16)ka; pk.h[1] = (_Float16)kb;
        *reinterpret_cast<unsigned int*>(&aLDS[u * AROW + 2 * kp]) = pk.w;
        pk.h[0] = (_Float16)kc; pk.h[1] = (_Float16)kd;
        *reinterpret_cast<unsigned int*>(&bLDS[u * AROW + 2 * kp]) = pk.w;
      }
    }
    __syncthreads();

    // ---- Phase 3: WMMA outer-product accumulation (EXEC all-ones) ----
    {
      const int m  = lane & 15;         // row within tile
      const int kh = lane >> 4;         // lane-half selects K group
#pragma unroll
      for (int s = 0; s < CHUNK / 32; ++s) {
        // A fragment (16x32 f16): lanes 0-15 K=0..15, lanes 16-31 K=16..31
        const _Float16* ap = &aLDS[(uT * 16 + m) * AROW + s * 32 + 16 * kh];
        const v8h alo = *reinterpret_cast<const v8h*>(ap);        // K = 16kh..16kh+7
        const v8h ahi = *reinterpret_cast<const v8h*>(ap + 8);    // K = 16kh+8..
        v16h a;
#pragma unroll
        for (int i = 0; i < 8; ++i) { a[i] = alo[i]; a[8 + i] = ahi[i]; }
        // B fragments (32x16 f16): col = lane%16, K = 16*kh .. 16*kh+15
        const v16h b0 = *reinterpret_cast<const v16h*>(
            &bLDS[(vT0 * 16 + m) * AROW + s * 32 + 16 * kh]);
        const v16h b1 = *reinterpret_cast<const v16h*>(
            &bLDS[((vT0 + 1) * 16 + m) * AROW + s * 32 + 16 * kh]);
        acc0 = __builtin_amdgcn_wmma_f32_16x16x32_f16(
            false, a, false, b0, (short)0, acc0, false, false);
        acc1 = __builtin_amdgcn_wmma_f32_16x16x32_f16(
            false, a, false, b1, (short)0, acc1, false, false);
      }
    }
    __syncthreads();
  }

  // ---- Writeout: C/D layout is VGPR r -> M = r + 8*(lane>>4), N = lane&15 ----
  float* __restrict__ o = out + ((size_t)(l * CHN + c)) * HB * HB;
  const int n   = lane & 15;
  const int mhi = (lane >> 4) << 3;
#pragma unroll
  for (int r = 0; r < 8; ++r) {
    const int u = uT * 16 + mhi + r;
    atomicAdd(&o[u * HB + vT0 * 16 + n],       acc0[r]);
    atomicAdd(&o[u * HB + (vT0 + 1) * 16 + n], acc1[r]);
  }
}

// ---------------------------------------------------------------------------
__global__ void zero_kernel(float* __restrict__ p, int n) {
  const int i = blockIdx.x * blockDim.x + threadIdx.x;
  if (i < n) p[i] = 0.0f;
}

// Per-sample normalization: out[l] /= (sum(out[l]) + eps)
__global__ __launch_bounds__(256)
void norm_kernel(float* __restrict__ out) {
  __shared__ float red[256];
  const int l   = blockIdx.x;
  const int tid = threadIdx.x;
  float* p = out + (size_t)l * CHN * HB * HB;
  float s = 0.0f;
  for (int i = tid; i < CHN * HB * HB; i += 256) s += p[i];
  red[tid] = s;
  __syncthreads();
  for (int w = 128; w > 0; w >>= 1) {
    if (tid < w) red[tid] += red[tid + w];
    __syncthreads();
  }
  const float scale = 1.0f / (red[0] + 1e-6f);
  for (int i = tid; i < CHN * HB * HB; i += 256) p[i] *= scale;
}

// ---------------------------------------------------------------------------
extern "C" void kernel_launch(void* const* d_in, const int* in_sizes, int n_in,
                              void* d_out, int out_size, void* d_ws, size_t ws_size,
                              hipStream_t stream) {
  (void)in_sizes; (void)n_in; (void)d_ws; (void)ws_size;
  const float* x = (const float*)d_in[0];
  float* out = (float*)d_out;

  // d_out is poisoned by the harness; zero it before atomic accumulation.
  zero_kernel<<<(out_size + 255) / 256, 256, 0, stream>>>(out, out_size);

  dim3 grid(SPLITS, CHN, L_SAMP);     // 384 workgroups of 8 wave32 waves
  hist_wmma_kernel<<<grid, 256, 0, stream>>>(x, out);

  norm_kernel<<<L_SAMP, 256, 0, stream>>>(out);
}
